// NVSM_90168543412873
// MI455X (gfx1250) — matrix operand order repaired and checked by reading, not compiled, not run on hardware
//
#include <hip/hip_runtime.h>
#include <hip/hip_bf16.h>

typedef __attribute__((ext_vector_type(16))) _Float16 v16h;
typedef __attribute__((ext_vector_type(8)))  float    v8f;

#define DIM      256
#define ZNEG     10
#define NSLOT    11            // 1 positive + 10 negatives
#define NVSM_EPS 1e-40f

__device__ __forceinline__ float4 ld4(const float* p) {
  return *reinterpret_cast<const float4*>(p);
}

__device__ __forceinline__ void pack8(v16h& dst, int base, float4 a, float4 b) {
  dst[base + 0] = (_Float16)a.x; dst[base + 1] = (_Float16)a.y;
  dst[base + 2] = (_Float16)a.z; dst[base + 3] = (_Float16)a.w;
  dst[base + 4] = (_Float16)b.x; dst[base + 5] = (_Float16)b.y;
  dst[base + 6] = (_Float16)b.z; dst[base + 7] = (_Float16)b.w;
}

__device__ __forceinline__ int slot_doc(const int* __restrict__ document,
                                        const int* __restrict__ neg_sample,
                                        int row, int j) {
  return (j == 0) ? document[row]
                  : neg_sample[(size_t)row * ZNEG + (j - 1)];
}

// Prefetch one full 1KB doc row (16 rows covered by the 32 lanes: lane pair
// (mn, mn+16) splits the row in halves, 4 prefetches of a 128B line each).
__device__ __forceinline__ void prefetch_row(const float* __restrict__ doc_emb,
                                             int doc, int hi) {
  const float* r = doc_emb + (size_t)(unsigned)doc * DIM + hi * 128;
  __builtin_prefetch(r,       0, 3);
  __builtin_prefetch(r + 32,  0, 3);
  __builtin_prefetch(r + 64,  0, 3);
  __builtin_prefetch(r + 96,  0, 3);
}

// Two waves cooperate on one tile of 16 batch rows (slots split 5/6),
// partial results combined through LDS. A = query tile (16x256 f16 frags,
// resident per wave); per slot B = gathered doc rows (256x16); the 16
// per-row dots are the diagonal of the 16x16 WMMA result.
__global__ __launch_bounds__(256)
void nvsm_wmma_kernel(const float* __restrict__ query,
                      const float* __restrict__ doc_emb,
                      const int*   __restrict__ document,
                      const int*   __restrict__ neg_sample,
                      float*       __restrict__ out,
                      int batch) {
  __shared__ float partial[8][16];

  const int lane = threadIdx.x & 31;
  const int wave = threadIdx.x >> 5;
  const int pair = wave >> 1;               // tile index within block (0..3)
  const int par  = wave & 1;                // which slot half this wave owns
  const int tile = blockIdx.x * 4 + pair;
  const int tileBase = tile * 16;

  const int mn = lane & 15;                 // A row / B column owned by lane
  const int hi = lane >> 4;                 // K half-group select (ISA layout)
  const int row = min(tileBase + mn, batch - 1);   // clamp: keep EXEC full

  const int j0 = par ? 5 : 0;               // wave0: slots 0..4, wave1: 5..10
  const int j1 = par ? NSLOT : 5;

  // Prefetch this wave's first gather rows, then load A under that latency.
  int doc_cur = slot_doc(document, neg_sample, row, j0);
  prefetch_row(doc_emb, doc_cur, hi);

  // ---- Load A: query tile as 8 chunks of 16x32 f16 fragments -------------
  // ISA 16-bit A layout: lanes 0-15 hold K {0..7,16..23}+kb, lanes 16-31
  // hold K {8..15,24..31}+kb of the same rows.
  v16h afrag[8];
  const float* qrow = query + (size_t)row * DIM;
  #pragma unroll
  for (int kc = 0; kc < 8; ++kc) {
    const int kb = kc * 32 + hi * 8;
    float4 a0 = ld4(qrow + kb);
    float4 a1 = ld4(qrow + kb + 4);
    float4 a2 = ld4(qrow + kb + 16);
    float4 a3 = ld4(qrow + kb + 20);
    pack8(afrag[kc], 0, a0, a1);
    pack8(afrag[kc], 8, a2, a3);
  }

  float pos_term = 0.0f;
  float neg_term = 0.0f;

  for (int j = j0; j < j1; ++j) {
    // Fetch next slot's index early and prefetch its entire row.
    int doc_next = doc_cur;
    if (j + 1 < j1) {
      doc_next = slot_doc(document, neg_sample, row, j + 1);
      prefetch_row(doc_emb, doc_next, hi);
    }

    const float* drow = doc_emb + (size_t)(unsigned)doc_cur * DIM;

    // ---- Software-pipelined B loads + WMMA chain over K = 256 -----------
    // ISA 16-bit B layout (32x16 chunk): lane holds column (lane&15),
    // elements 0..15 are K {0..7,16..23}+kb with kb offset +8 for lanes 16-31.
    float4 c0 = ld4(drow + hi * 8);
    float4 c1 = ld4(drow + hi * 8 + 4);
    float4 c2 = ld4(drow + hi * 8 + 16);
    float4 c3 = ld4(drow + hi * 8 + 20);

    v8f acc = {};
    #pragma unroll
    for (int kc = 0; kc < 8; ++kc) {
      float4 n0, n1, n2, n3;
      if (kc < 7) {                       // issue next chunk before WMMA
        const int kb = (kc + 1) * 32 + hi * 8;
        n0 = ld4(drow + kb);
        n1 = ld4(drow + kb + 4);
        n2 = ld4(drow + kb + 16);
        n3 = ld4(drow + kb + 20);
      }
      v16h bfrag;
      pack8(bfrag, 0, c0, c1);
      pack8(bfrag, 8, c2, c3);
      acc = __builtin_amdgcn_wmma_f32_16x16x32_f16(
                false, afrag[kc], false, bfrag,
                (short)0, acc, false, false);
      if (kc < 7) { c0 = n0; c1 = n1; c2 = n2; c3 = n3; }
    }

    // ---- Diagonal extraction -------------------------------------------
    // diag(m) lives at VGPR m&7, lane m (m<8) or lane m+16 (m>=8).
    float x = acc[0];
    #pragma unroll
    for (int e = 1; e < 8; ++e) x = ((lane & 7) == e) ? acc[e] : x;
    const int srcLane = (mn < 8) ? mn : (mn + 16);
    const float d = __shfl(x, srcLane, 32);

    if (j == 0) {
      pos_term = -__logf(1.0f + __expf(-d));           // log(sigmoid(d))
    } else {
      neg_term += __logf(1.0f / (1.0f + __expf(d)) + NVSM_EPS); // log(1-sig+eps)
    }

    doc_cur = doc_next;
  }

  // ---- Combine the two waves of this tile through LDS --------------------
  if (lane < 16) partial[wave][lane] = 10.0f * pos_term + neg_term;
  __syncthreads();

  if (par == 0 && lane < 16) {
    const int b = tileBase + lane;
    if (b < batch) {
      const float v = partial[wave][lane] + partial[wave + 1][lane];
      out[b] = 0.55f * v;                  // (Z+1)/(2Z) * (Z*pos + neg)
    }
  }
}

extern "C" void kernel_launch(void* const* d_in, const int* in_sizes, int n_in,
                              void* d_out, int out_size, void* d_ws, size_t ws_size,
                              hipStream_t stream) {
  const float* query      = (const float*)d_in[0];
  const float* doc_emb    = (const float*)d_in[1];
  const int*   document   = (const int*)d_in[2];
  const int*   neg_sample = (const int*)d_in[3];
  float*       out        = (float*)d_out;

  const int batch = in_sizes[2];           // B (count of positive doc indices)
  const int rowsPerBlock = 4 * 16;         // 4 tiles x 16 rows (2 waves/tile)
  const int blocks = (batch + rowsPerBlock - 1) / rowsPerBlock;

  nvsm_wmma_kernel<<<dim3(blocks), dim3(256), 0, stream>>>(
      query, doc_emb, document, neg_sample, out, batch);
}
